// NetworkBasic_43997644980495
// MI455X (gfx1250) — matrix operand order repaired and checked by reading, not compiled, not run on hardware
//
#include <hip/hip_runtime.h>
#include <hip/hip_bf16.h>

typedef __attribute__((ext_vector_type(16))) _Float16     v16h;
typedef __attribute__((ext_vector_type(8)))  float        v8f;
typedef __attribute__((ext_vector_type(4)))  unsigned int v4u;
typedef __attribute__((ext_vector_type(8)))  int          v8i;
typedef __attribute__((ext_vector_type(4)))  int          v4i;

#define TT    64      // time steps
#define TILE  16      // interior tile edge
#define HALO  18      // TILE + 2
#define NP    (HALO*HALO)   // 324 pixels incl. halo
#define NGRP  21            // ceil(324/16)
#define NPAD  (NGRP*16)     // 336 padded rows
#define BATCH 16
#define HH    128
#define WW    64

// HAVE_TDM: device pass -> real probe; host pass -> 1 (parse only, stub declared below).
#if defined(__HIP_DEVICE_COMPILE__)
#  if __has_builtin(__builtin_amdgcn_tensor_load_to_lds)
#    define HAVE_TDM 1
#  else
#    define HAVE_TDM 0
#  endif
#else
#  define HAVE_TDM 1
#endif

// ---- constant tables (match reference _alpha_kernel cutoffs exactly) ----
__device__ const float d_srm1[8] = {
  0.0f, 1.0f, 0.73575888f, 0.40600585f, 0.19914827f, 0.09157819f, 0.04042768f, 0.01735127f };
__device__ const float d_srm2[16] = {
  0.0f, 0.82436064f, 1.0f, 0.90979599f, 0.73575888f, 0.55782540f, 0.40600585f, 0.28729750f,
  0.19914827f, 0.13588823f, 0.09157819f, 0.06109948f, 0.04042768f, 0.02656401f,
  0.01735127f, 0.01127579f };
__device__ const float d_ref1[18] = {
  0.0f, -60.0f, -44.14553294f, -24.36035098f, -11.94889641f, -5.49469166f, -2.42566091f,
  -1.04107591f, -0.43770334f, -0.18114982f, -0.07404588f, -0.02996395f, -0.01202522f,
  -0.00479249f, -0.00189868f, -0.00074838f, -0.00029367f, -0.00011479f };
__device__ const float d_ref2[36] = {
  0.0f, -82.43606354f, -100.0f, -90.97959896f, -73.57588823f, -55.78254003f, -40.60058497f,
  -28.72974951f, -19.91482735f, -13.58882253f, -9.15781944f, -6.10994808f, -4.04276819f,
  -2.65640141f, -1.73512652f, -1.12757933f, -0.72950557f, -0.47012173f, -0.30191637f,
  -0.19329497f, -0.12340980f, -0.07859442f, -0.04993992f, -0.03166692f, -0.02004204f,
  -0.01266262f, -0.00798748f, -0.00503098f, -0.00316446f, -0.00198789f, -0.00124729f,
  -0.00078174f, -0.00048944f, -0.00030614f, -0.00019131f, -0.00011945f };

// Issue one 1-D TDM row copy: nelem f16 elements, global -> LDS.
// Always declared (host pass parses discarded constexpr branches); body is device-only.
__device__ __forceinline__ void tdm_row_to_lds(unsigned long long gaddr, unsigned lds_addr,
                                               unsigned nelem)
{
#if defined(__HIP_DEVICE_COMPILE__) && HAVE_TDM
    v4u g0;
    g0[0] = 1u;                                               // count=1 (valid user D#)
    g0[1] = lds_addr;                                         // lds_addr (bytes)
    g0[2] = (unsigned)(gaddr & 0xFFFFFFFFull);                // global_addr[31:0]
    g0[3] = (unsigned)((gaddr >> 32) & 0x1FFFFFFull)          // global_addr[56:32]
          | (2u << 30);                                       // type=2 ("image")
    v8i g1;
    g1[0] = 0x00010000;                                       // wg_mask=0, data_size=1 (2B)
    g1[1] = (int)((nelem & 0xFFFFu) << 16);                   // tensor_dim0[15:0]
    g1[2] = (int)((nelem >> 16) | (1u << 16));                // tensor_dim0[31:16], tensor_dim1=1
    g1[3] = (int)(nelem << 16);                               // tensor_dim1 hi=0, tile_dim0=nelem
    g1[4] = 1;                                                // tile_dim1=1, tile_dim2=0
    g1[5] = (int)nelem;                                       // tensor_dim0_stride lo (1 row)
    g1[6] = 0;                                                // stride hi / dim1_stride lo
    g1[7] = 0;                                                // dim1_stride hi
    v4i gz = {0, 0, 0, 0};
#if defined(__clang_major__) && __clang_major__ >= 23
    v8i gz8 = {0, 0, 0, 0, 0, 0, 0, 0};
    __builtin_amdgcn_tensor_load_to_lds(g0, g1, gz, gz, gz8, 0);
#else
    __builtin_amdgcn_tensor_load_to_lds(g0, g1, gz, gz, 0);
#endif
#else
    (void)gaddr; (void)lds_addr; (void)nelem;
#endif
}

// One fused SNN layer: [load spikes tile+halo] -> [PSP via WMMA, in-place in LDS]
// -> [3x3 spatial conv + sequential spike scan] -> store spikes.
template<int SRMLEN, int REFLEN, bool IN_F16, bool OUT_F16>
__global__ __launch_bounds__(256) void snn_layer(const void* __restrict__ inp,
                                                 const float* __restrict__ w9,
                                                 void* __restrict__ outp,
                                                 float theta)
{
    __shared__ __align__(16) _Float16 XP[NPAD * TT];  // spike series -> PSP (in place)
    __shared__ __align__(16) _Float16 MK[TT * TT];    // causal temporal matrix M[t_in][t_out]

    const int tid  = threadIdx.x;
    const int bw   = blockIdx.x * TILE;   // base w
    const int bh   = blockIdx.y * TILE;   // base h
    const int bb   = blockIdx.z;          // batch
    const int wave = tid >> 5;
    const int lane = tid & 31;

    // ---- Phase 0: fill LDS with input spike series (f16), zero-pad halo/pad rows ----
    if constexpr (IN_F16 && HAVE_TDM) {
        // Zero all of XP (covers out-of-image halo and WMMA pad rows), 16B per store.
        for (int idx = tid; idx < (NPAD * TT * 2) / 16; idx += 256)
            ((v4i*)XP)[idx] = (v4i){0, 0, 0, 0};
        __syncthreads();
        if (wave == 0) {
            // Clamp halo tile to the image; one TDM row per in-bounds h line.
            const int x0   = (bw - 1 < 0) ? 0 : bw - 1;
            const int x1   = (bw + HALO - 1 > WW) ? WW : bw + HALO - 1;
            const int lead = x0 - (bw - 1);                   // 0 or 1
            const unsigned nelem = (unsigned)(x1 - x0) * TT;  // f16 elements per row
            const unsigned long long gbase = (unsigned long long)(uintptr_t)inp;
            const unsigned ldsbase = (unsigned)(uintptr_t)XP;
            for (int hy = 0; hy < HALO; ++hy) {
                const int h = bh + hy - 1;
                if (h < 0 || h >= HH) continue;               // stays zero
                const unsigned long long ga =
                    gbase + ((((unsigned long long)bb * HH + h) * WW + x0) * TT) * 2ull;
                const unsigned la = ldsbase + (unsigned)((hy * HALO + lead) * TT) * 2u;
                tdm_row_to_lds(ga, la, nelem);
            }
        }
    } else {
        for (int idx = tid; idx < NPAD * TT; idx += 256) {
            const int hp = idx >> 6, t = idx & 63;
            float v = 0.0f;
            if (hp < NP) {
                const int hy = hp / HALO, hx = hp - hy * HALO;
                const int h = bh + hy - 1, w = bw + hx - 1;
                if (h >= 0 && h < HH && w >= 0 && w < WW) {
                    const size_t off = ((((size_t)bb * HH + h) * WW + w) * TT) + t;
                    v = IN_F16 ? (float)((const _Float16*)inp)[off] : ((const float*)inp)[off];
                }
            }
            XP[idx] = (_Float16)v;
        }
    }
    // MK fill overlaps the TDM DMA.
    const float* srm = (SRMLEN == 8) ? d_srm1 : d_srm2;
    for (int idx = tid; idx < TT * TT; idx += 256) {
        const int tin = idx >> 6, tout = idx & 63;
        const int d = tout - tin;
        MK[idx] = (_Float16)((d >= 0 && d < SRMLEN) ? srm[d] : 0.0f);
    }
    if constexpr (IN_F16 && HAVE_TDM) {
        if (wave == 0) __builtin_amdgcn_s_wait_tensorcnt(0);
    }
    __syncthreads();

    // ---- Phase 1: PSP temporal conv = (16 px x 64 t) @ (64 x 64) via v_wmma_f32_16x16x32_f16 ----
    const int hhalf = lane >> 4;   // lane half select
    const int lm    = lane & 15;

    for (int g = wave; g < NGRP; g += 8) {
        const int pg = g * 16;
        // A fragments (16x32 f16) for K-chunks c=0 (t_in 0..31) and c=1 (t_in 32..63)
        v16h a0, a1;
        const _Float16* xrow = &XP[(pg + lm) * TT];
        #pragma unroll
        for (int v = 0; v < 8; ++v) {
            const int kl = ((v >> 2) * 16) + hhalf * 8 + 2 * (v & 3);
            a0[2*v]   = xrow[kl];
            a0[2*v+1] = xrow[kl + 1];
            a1[2*v]   = xrow[32 + kl];
            a1[2*v+1] = xrow[32 + kl + 1];
        }
        v8f accv[4];
        #pragma unroll
        for (int n = 0; n < 4; ++n) {
            // B fragments (32x16 f16): rows = t_in, cols = t_out tile n
            v16h b0, b1;
            #pragma unroll
            for (int v = 0; v < 8; ++v) {
                const int kl  = ((v >> 2) * 16) + hhalf * 8 + 2 * (v & 3);
                const int col = n * 16 + lm;
                b0[2*v]   = MK[(kl)     * TT + col];
                b0[2*v+1] = MK[(kl + 1) * TT + col];
                b1[2*v]   = MK[(32 + kl)     * TT + col];
                b1[2*v+1] = MK[(32 + kl + 1) * TT + col];
            }
            v8f c = {};
            c = __builtin_amdgcn_wmma_f32_16x16x32_f16(false, a0, false, b0, (short)0, c, false, false);
            c = __builtin_amdgcn_wmma_f32_16x16x32_f16(false, a1, false, b1, (short)0, c, false, false);
            accv[n] = c;
        }
        // Write PSP back in place (this wave owns rows pg..pg+15 exclusively)
        #pragma unroll
        for (int n = 0; n < 4; ++n) {
            #pragma unroll
            for (int r = 0; r < 8; ++r) {
                XP[(pg + r + 8 * hhalf) * TT + n * 16 + lm] = (_Float16)accv[n][r];
            }
        }
    }
    __syncthreads();

    // ---- Phase 2: 3x3 spatial conv (zero-pad) + sequential refractory spike scan ----
    // Shift-register refractory buffer: buf[0] is the contribution landing at current t.
    const int iy = tid >> 4, ix = tid & 15;   // interior pixel
    float w[9];
    #pragma unroll
    for (int i = 0; i < 9; ++i) w[i] = w9[i];

    const float* reft = (REFLEN == 18) ? d_ref1 : d_ref2;
    float rk[REFLEN];
    #pragma unroll
    for (int j = 0; j < REFLEN; ++j) rk[j] = reft[j];

    float buf[REFLEN];
    #pragma unroll
    for (int j = 0; j < REFLEN; ++j) buf[j] = 0.0f;

    const _Float16* xc = &XP[(iy * HALO + ix) * TT];
    const size_t obase = (((size_t)bb * HH + (bh + iy)) * WW + (bw + ix)) * TT;

    for (int t = 0; t < TT; ++t) {   // rolled loop; all per-step arrays statically indexed
        float u = 0.0f;
        #pragma unroll
        for (int ky = 0; ky < 3; ++ky) {
            #pragma unroll
            for (int kx = 0; kx < 3; ++kx) {
                u = fmaf(w[ky * 3 + kx], (float)xc[(ky * HALO + kx) * TT + t], u);
            }
        }
        const float m  = u + buf[0];
        const float sp = (m >= theta) ? 1.0f : 0.0f;
        #pragma unroll
        for (int j = 0; j < REFLEN - 1; ++j) buf[j] = fmaf(sp, rk[j + 1], buf[j + 1]);
        buf[REFLEN - 1] = 0.0f;
        if (OUT_F16) ((_Float16*)outp)[obase + t] = (_Float16)sp;
        else         ((float*)outp)[obase + t]    = sp;
    }
}

extern "C" void kernel_launch(void* const* d_in, const int* in_sizes, int n_in,
                              void* d_out, int out_size, void* d_ws, size_t ws_size,
                              hipStream_t stream) {
    (void)in_sizes; (void)n_in; (void)out_size; (void)ws_size;
    const void*  spikeIn = d_in[0];                  // f32 [16,1,128,64,64]
    const float* w1      = (const float*)d_in[1];    // 9 f32
    const float* w2      = (const float*)d_in[2];    // 9 f32
    void*        s1      = d_ws;                     // f16 [16,128,64,64] = 16 MB scratch

    dim3 grid(WW / TILE, HH / TILE, BATCH);          // (4, 8, 16)
    dim3 block(256);

    // Layer 1: spikes(f32) -> s1 spikes(f16 in ws), theta=30, srm tau=1, ref len 18
    snn_layer<8, 18, false, true><<<grid, block, 0, stream>>>(spikeIn, w1, s1, 30.0f);
    // Layer 2: s1(f16, TDM tile fill) -> s2(f32 out), theta=50, srm tau=2, ref len 36
    snn_layer<16, 36, true, false><<<grid, block, 0, stream>>>(s1, w2, d_out, 50.0f);
}
